// DMPNNModel_32744830665394
// MI455X (gfx1250) — matrix operand order repaired and checked by reading, not compiled, not run on hardware
//
#include <hip/hip_runtime.h>
#include <hip/hip_bf16.h>

// ---------------------------------------------------------------------------
// DMPNN on MI455X (gfx1250): bf16 WMMA GEMMs (pre-packed bf16 weights,
// bf16 LDS A-tiles -> zero-convert inner loop) + L2-resident f32 atomics
// ---------------------------------------------------------------------------

#define NN 50000
#define EE 600000
#define HH 128
#define ECC 8
#define LL 3
#define GG 512

typedef __attribute__((ext_vector_type(16))) __bf16       v16bf;
typedef __attribute__((ext_vector_type(8)))  float        v8f;
typedef __attribute__((ext_vector_type(4)))  unsigned int v4u;

union frag_u { v4u q[2]; v16bf v; };
union pk_u   { __bf16 h[2]; unsigned int u; };

// One wave computes a 16x16 tile of (A_tile @ W^T).
// AsU: LDS A tile, bf16 packed 2-per-uint, [16][kpad] logical bf16 layout.
// Wp : global bf16 weights packed 2-per-uint, row-major [Nout][kpad], zero-padded.
// Fragment layouts per CDNA5 ISA:
//   A: lane holds row m=lane&15; elems e -> k = (lane&16?8:0) + (e<8?e:e+8)
//      => two contiguous 8xbf16 runs at k0 and k0+16  => two 16B LDS loads
//   B: lane holds col n=lane&15; elems e -> k = (lane&16?8:0) + (e<8?e:e+8)
//      => two contiguous 8xbf16 runs at k0 and k0+16 => two 16B global loads
__device__ __forceinline__ v8f wmma_rowtile(const unsigned int* AsU, int kpad,
                                            const unsigned int* Wp, int n, int lane)
{
    v8f acc = {0.f, 0.f, 0.f, 0.f, 0.f, 0.f, 0.f, 0.f};
    const int m    = lane & 15;
    const int aoff = (lane & 16) ? 8  : 0;
    const int boff = (lane & 16) ? 8  : 0;
    const unsigned int* wrow = Wp + (((size_t)n * kpad) >> 1);
    for (int kk = 0; kk < kpad; kk += 32) {
        frag_u ua, ub;
        const v4u* pa = (const v4u*)(AsU + ((m * kpad + kk + aoff) >> 1));
        ua.q[0] = pa[0];
        ua.q[1] = pa[2];           // +16 bf16 = +2 x 16B
        const v4u* pb = (const v4u*)(wrow + ((kk + boff) >> 1));
        ub.q[0] = pb[0];
        ub.q[1] = pb[2];
        acc = __builtin_amdgcn_wmma_f32_16x16x32_bf16(false, ua.v, false, ub.v,
                                                      (short)0, acc, false, false);
    }
    return acc;
}

__device__ __forceinline__ unsigned int pack2(float a, float b)
{
    pk_u p;
    p.h[0] = (__bf16)a;
    p.h[1] = (__bf16)b;
    return p.u;
}

// ---- weight pre-pack: f32 [Nout,K] -> bf16x2 [Nout,Kpad/2], zero-padded ----
__global__ void k_pack(const float* __restrict__ W, unsigned int* __restrict__ Wp,
                       int K, int Kpad, int total)
{
    int i = blockIdx.x * 256 + threadIdx.x;
    if (i >= total) return;
    int kp2 = Kpad >> 1;
    int n = i / kp2;
    int k = (i - n * kp2) * 2;
    float a = (k     < K) ? W[(size_t)n * K + k]     : 0.f;
    float b = (k + 1 < K) ? W[(size_t)n * K + k + 1] : 0.f;
    Wp[i] = pack2(a, b);
}

// ---------------- embed: h = relu(LN(x @ W^T + b)) -------------------------
__global__ void __launch_bounds__(256)
k_embed(const float* __restrict__ x, const unsigned int* __restrict__ Wp,
        const float* __restrict__ bias, const float* __restrict__ lng,
        const float* __restrict__ lnb, float* __restrict__ out)
{
    __shared__ __attribute__((aligned(16))) unsigned int AsU[16 * 128 / 2];
    __shared__ float Ot[16 * 128];
    __shared__ float s1[16][16];
    __shared__ float s2[16][16];
    __shared__ float red[32];
    const int rowBase = blockIdx.x * 16;
    const int t = threadIdx.x;
    for (int i = t; i < 16 * 64; i += 256) {
        int g = 2 * i;  // global flat f32 index within tile
        AsU[i] = pack2(x[(size_t)rowBase * 128 + g], x[(size_t)rowBase * 128 + g + 1]);
    }
    __syncthreads();
    const int wave = t >> 5, lane = t & 31;
    const int n = wave * 16 + (lane & 15);
    v8f acc = wmma_rowtile(AsU, 128, Wp, n, lane);
    const int rhi = (lane & 16) ? 8 : 0;
#pragma unroll
    for (int r = 0; r < 8; ++r)
        Ot[(r + rhi) * 128 + n] = acc[r] + bias[n];
    __syncthreads();
    const int rr = t >> 4, cc = t & 15;
    float s = 0.f, q = 0.f;
    for (int c = cc; c < 128; c += 16) { float v = Ot[rr * 128 + c]; s += v; q += v * v; }
    s1[rr][cc] = s; s2[rr][cc] = q;
    __syncthreads();
    if (cc == 0) {
        float S = 0.f, Q = 0.f;
        for (int i = 0; i < 16; ++i) { S += s1[rr][i]; Q += s2[rr][i]; }
        float mean = S * (1.f / 128.f);
        float var  = Q * (1.f / 128.f) - mean * mean;
        red[rr * 2]     = mean;
        red[rr * 2 + 1] = rsqrtf(var + 1e-5f);
    }
    __syncthreads();
    for (int i = t; i < 16 * 128; i += 256) {
        int r = i >> 7, c = i & 127;
        float v = (Ot[i] - red[r * 2]) * red[r * 2 + 1] * lng[c] + lnb[c];
        out[(size_t)rowBase * 128 + i] = fmaxf(v, 0.f);
    }
}

// -------- msgs init: msgs = relu(concat(h[src], edge_attr) @ Wi^T) ---------
__global__ void __launch_bounds__(256)
k_msg_init(const float* __restrict__ h, const float* __restrict__ ea,
           const int* __restrict__ src, const unsigned int* __restrict__ WiP,
           float* __restrict__ msgs)
{
    __shared__ __attribute__((aligned(16))) unsigned int AsU[16 * 160 / 2];
    const int rowBase = blockIdx.x * 16;
    const int t = threadIdx.x;
    for (int i = t; i < 16 * 80; i += 256) {
        int r = i / 80, k = (i - r * 80) * 2;     // pairs never straddle 128/136
        int e = rowBase + r;
        float a = 0.f, b = 0.f;
        if (k < 128) {
            const float* hp = h + (size_t)src[e] * 128 + k;
            a = hp[0]; b = hp[1];
        } else if (k < 136) {
            const float* ep = ea + (size_t)e * 8 + (k - 128);
            a = ep[0]; b = ep[1];
        }
        AsU[i] = pack2(a, b);
    }
    __syncthreads();
    const int wave = t >> 5, lane = t & 31;
    const int n = wave * 16 + (lane & 15);
    v8f acc = wmma_rowtile(AsU, 160, WiP, n, lane);
    const int rhi = (lane & 16) ? 8 : 0;
#pragma unroll
    for (int r = 0; r < 8; ++r)
        msgs[(size_t)(rowBase + r + rhi) * 128 + n] = fmaxf(acc[r], 0.f);
}

// ---- edge update: msgs = LN(relu(agg[src] @ Wh^T) + msgs)  (in place) -----
__global__ void __launch_bounds__(256)
k_edge(const float* __restrict__ agg, const int* __restrict__ src,
       const unsigned int* __restrict__ WhP, const float* __restrict__ lng,
       const float* __restrict__ lnb, float* __restrict__ msgs)
{
    __shared__ __attribute__((aligned(16))) unsigned int AsU[16 * 128 / 2];
    __shared__ float Ot[16 * 128];
    __shared__ float s1[16][16];
    __shared__ float s2[16][16];
    __shared__ float red[32];
    const int rowBase = blockIdx.x * 16;
    const int t = threadIdx.x;
    for (int i = t; i < 16 * 64; i += 256) {
        int r = i >> 6, k = (i & 63) * 2;
        const float* ap = agg + (size_t)src[rowBase + r] * 128 + k;
        AsU[i] = pack2(ap[0], ap[1]);
    }
    __syncthreads();
    const int wave = t >> 5, lane = t & 31;
    const int n = wave * 16 + (lane & 15);
    v8f acc = wmma_rowtile(AsU, 128, WhP, n, lane);
    const int rhi = (lane & 16) ? 8 : 0;
#pragma unroll
    for (int r = 0; r < 8; ++r) {
        int row = rowBase + r + rhi;
        Ot[(r + rhi) * 128 + n] = fmaxf(acc[r], 0.f) + msgs[(size_t)row * 128 + n];
    }
    __syncthreads();
    const int rr = t >> 4, cc = t & 15;
    float s = 0.f, q = 0.f;
    for (int c = cc; c < 128; c += 16) { float v = Ot[rr * 128 + c]; s += v; q += v * v; }
    s1[rr][cc] = s; s2[rr][cc] = q;
    __syncthreads();
    if (cc == 0) {
        float S = 0.f, Q = 0.f;
        for (int i = 0; i < 16; ++i) { S += s1[rr][i]; Q += s2[rr][i]; }
        float mean = S * (1.f / 128.f);
        float var  = Q * (1.f / 128.f) - mean * mean;
        red[rr * 2]     = mean;
        red[rr * 2 + 1] = rsqrtf(var + 1e-5f);
    }
    __syncthreads();
    for (int i = t; i < 16 * 128; i += 256) {
        int r = i >> 7, c = i & 127;
        float v = (Ot[i] - red[r * 2]) * red[r * 2 + 1] * lng[c] + lnb[c];
        msgs[(size_t)rowBase * 128 + i] = v;
    }
}

// ---- node update: h = bn(relu(concat(h, node_msg) @ Wo^T + bo)) [opt relu] -
__global__ void __launch_bounds__(256)
k_node(const float* __restrict__ hin, const float* __restrict__ nm,
       const unsigned int* __restrict__ WoP, const float* __restrict__ bo,
       const float* __restrict__ bng, const float* __restrict__ bnb,
       int relu_out, float* __restrict__ hout)
{
    __shared__ __attribute__((aligned(16))) unsigned int AsU[16 * 256 / 2];
    const int rowBase = blockIdx.x * 16;
    const int t = threadIdx.x;
    for (int i = t; i < 16 * 128; i += 256) {
        int r = i >> 7, k = (i & 127) * 2;        // pairs never straddle 128
        const float* p = (k < 128) ? (hin + (size_t)(rowBase + r) * 128 + k)
                                   : (nm  + (size_t)(rowBase + r) * 128 + (k - 128));
        AsU[i] = pack2(p[0], p[1]);
    }
    __syncthreads();
    const int wave = t >> 5, lane = t & 31;
    const int n = wave * 16 + (lane & 15);
    v8f acc = wmma_rowtile(AsU, 256, WoP, n, lane);
    const int rhi = (lane & 16) ? 8 : 0;
    const float bnscale = 0.999997500006250f;  // ~1/sqrt(1+5e-6)
#pragma unroll
    for (int r = 0; r < 8; ++r) {
        float v = fmaxf(acc[r] + bo[n], 0.f);
        v = v * (bng[n] * bnscale) + bnb[n];
        if (relu_out) v = fmaxf(v, 0.f);
        hout[(size_t)(rowBase + r + rhi) * 128 + n] = v;
    }
}

// ---------------- scatter / pooling / classifier helpers -------------------
__global__ void k_clear(float* __restrict__ p, int n)
{
    int i = blockIdx.x * 256 + threadIdx.x;
    if (i < n) p[i] = 0.f;
}

__global__ void k_scatter(const float* __restrict__ msgs, const int* __restrict__ dst,
                          float* __restrict__ agg, int total)
{
    int i = blockIdx.x * 256 + threadIdx.x;
    if (i >= total) return;
    int e = i >> 7, c = i & 127;
    unsafeAtomicAdd(&agg[(size_t)dst[e] * 128 + c], msgs[i]);
}

__global__ void k_pool(const float* __restrict__ h, const int* __restrict__ batch,
                       float* __restrict__ sums, float* __restrict__ counts, int total)
{
    int i = blockIdx.x * 256 + threadIdx.x;
    if (i >= total) return;
    int node = i >> 7, c = i & 127;
    int g = batch[node];
    unsafeAtomicAdd(&sums[(size_t)g * 128 + c], h[i]);
    if (c == 0) unsafeAtomicAdd(&counts[g], 1.0f);
}

__global__ void k_div(const float* __restrict__ sums, const float* __restrict__ counts,
                      float* __restrict__ gemb)
{
    int i = blockIdx.x * 256 + threadIdx.x;
    if (i >= GG * 128) return;
    gemb[i] = sums[i] / fmaxf(counts[i >> 7], 1.0f);
}

// classifier layer 1: hid = relu(gemb @ W1^T + b1), Nout = 64 (4 waves/block)
__global__ void __launch_bounds__(128)
k_cls1(const float* __restrict__ gemb, const unsigned int* __restrict__ W1P,
       const float* __restrict__ b1, float* __restrict__ hid)
{
    __shared__ __attribute__((aligned(16))) unsigned int AsU[16 * 128 / 2];
    const int rowBase = blockIdx.x * 16;
    const int t = threadIdx.x;
    for (int i = t; i < 16 * 64; i += 128) {
        int g = 2 * i;
        AsU[i] = pack2(gemb[(size_t)rowBase * 128 + g], gemb[(size_t)rowBase * 128 + g + 1]);
    }
    __syncthreads();
    const int wave = t >> 5, lane = t & 31;
    const int n = wave * 16 + (lane & 15);
    v8f acc = wmma_rowtile(AsU, 128, W1P, n, lane);
    const int rhi = (lane & 16) ? 8 : 0;
#pragma unroll
    for (int r = 0; r < 8; ++r)
        hid[(size_t)(rowBase + r + rhi) * 64 + n] = fmaxf(acc[r] + b1[n], 0.f);
}

// classifier layer 2: out = hid @ W2^T + b2, OUT = 2 (tiny -> plain VALU)
__global__ void k_cls2(const float* __restrict__ hid, const float* __restrict__ W2,
                       const float* __restrict__ b2, float* __restrict__ out)
{
    int i = blockIdx.x * 256 + threadIdx.x;
    if (i >= GG * 2) return;
    int g = i >> 1, o = i & 1;
    float s = b2[o];
    for (int c = 0; c < 64; ++c) s += hid[(size_t)g * 64 + c] * W2[o * 64 + c];
    out[i] = s;
}

// ---------------------------------------------------------------------------
extern "C" void kernel_launch(void* const* d_in, const int* in_sizes, int n_in,
                              void* d_out, int out_size, void* d_ws, size_t ws_size,
                              hipStream_t stream)
{
    const float* x        = (const float*)d_in[0];
    const int*   ei       = (const int*)  d_in[1];
    const float* ea       = (const float*)d_in[2];
    const int*   batch    = (const int*)  d_in[3];
    const float* emb_W    = (const float*)d_in[4];
    const float* emb_b    = (const float*)d_in[5];
    const float* emb_ln_g = (const float*)d_in[6];
    const float* emb_ln_b = (const float*)d_in[7];
    const float* Wi       = (const float*)d_in[8];    // [L,H,H+EC] (only l=0 used)
    const float* Wh       = (const float*)d_in[9];    // [L,H,H]
    const float* Wo       = (const float*)d_in[10];   // [L,H,2H]
    const float* bo       = (const float*)d_in[11];   // [L,H]
    const float* ln_g     = (const float*)d_in[12];
    const float* ln_b     = (const float*)d_in[13];
    const float* bn_g     = (const float*)d_in[14];
    const float* bn_b     = (const float*)d_in[15];
    const float* cls_W1   = (const float*)d_in[16];
    const float* cls_b1   = (const float*)d_in[17];
    const float* cls_W2   = (const float*)d_in[18];
    const float* cls_b2   = (const float*)d_in[19];
    float* out = (float*)d_out;

    const int* src = ei;
    const int* dst = ei + EE;

    char* ws = (char*)d_ws;
    auto alloc = [&](size_t bytes) -> char* {
        char* p = ws;
        ws += (bytes + 255) & ~(size_t)255;
        return p;
    };
    float* h    = (float*)alloc((size_t)NN * 128 * 4);
    float* msgs = (float*)alloc((size_t)EE * 128 * 4);
    float* agg  = (float*)alloc((size_t)NN * 128 * 4);
    float* nmsg = (float*)alloc((size_t)NN * 128 * 4);
    float* sums = (float*)alloc((size_t)GG * 128 * 4);
    float* cnts = (float*)alloc((size_t)GG * 4);
    float* gemb = (float*)alloc((size_t)GG * 128 * 4);
    float* hid  = (float*)alloc((size_t)GG * 64 * 4);
    // packed bf16 weights (2 per uint)
    unsigned int* embWp = (unsigned int*)alloc((size_t)128 * 128 * 2);
    unsigned int* WiP   = (unsigned int*)alloc((size_t)128 * 160 * 2);
    unsigned int* WhP   = (unsigned int*)alloc((size_t)3 * 128 * 128 * 2);
    unsigned int* WoP   = (unsigned int*)alloc((size_t)3 * 128 * 256 * 2);
    unsigned int* W1P   = (unsigned int*)alloc((size_t)64 * 128 * 2);

    // ---- pack weights to bf16 (zero-padded K) ----
    {
        int tot;
        tot = 128 * 128 / 2;       // emb_W: [128,128]
        k_pack<<<(tot + 255) / 256, 256, 0, stream>>>(emb_W, embWp, 128, 128, tot);
        tot = 128 * 160 / 2;       // Wi[0]: [128,136] -> pad 160
        k_pack<<<(tot + 255) / 256, 256, 0, stream>>>(Wi, WiP, 136, 160, tot);
        tot = 3 * 128 * 128 / 2;   // Wh: treat as [384,128]
        k_pack<<<(tot + 255) / 256, 256, 0, stream>>>(Wh, WhP, 128, 128, tot);
        tot = 3 * 128 * 256 / 2;   // Wo: treat as [384,256]
        k_pack<<<(tot + 255) / 256, 256, 0, stream>>>(Wo, WoP, 256, 256, tot);
        tot = 64 * 128 / 2;        // cls_W1: [64,128]
        k_pack<<<(tot + 255) / 256, 256, 0, stream>>>(cls_W1, W1P, 128, 128, tot);
    }

    const int nodeBlocks = NN / 16;      // 3125
    const int edgeBlocks = EE / 16;      // 37500
    const int scatN = EE * 128;          // 76.8M
    const int poolN = NN * 128;          // 6.4M

    // node embedding
    k_embed<<<nodeBlocks, 256, 0, stream>>>(x, embWp, emb_b, emb_ln_g, emb_ln_b, h);

    // initial edge messages (layer 0 input transform)
    k_msg_init<<<edgeBlocks, 256, 0, stream>>>(h, ea, src, WiP, msgs);

    for (int l = 0; l < LL; ++l) {
        const unsigned int* WhPl = WhP + (size_t)l * 128 * 128 / 2;
        const unsigned int* WoPl = WoP + (size_t)l * 128 * 256 / 2;

        // agg = segment_sum(msgs, dst)
        k_clear<<<(NN * 128 + 255) / 256, 256, 0, stream>>>(agg, NN * 128);
        k_scatter<<<(scatN + 255) / 256, 256, 0, stream>>>(msgs, dst, agg, scatN);

        // msgs = LN(relu(agg[src] @ Wh^T) + msgs)
        k_edge<<<edgeBlocks, 256, 0, stream>>>(agg, src, WhPl,
                                               ln_g + (size_t)l * HH,
                                               ln_b + (size_t)l * HH, msgs);

        // node_msg = segment_sum(msgs, dst)
        k_clear<<<(NN * 128 + 255) / 256, 256, 0, stream>>>(nmsg, NN * 128);
        k_scatter<<<(scatN + 255) / 256, 256, 0, stream>>>(msgs, dst, nmsg, scatN);

        // h = bn(relu(concat(h, node_msg) @ Wo^T + bo)) [relu if l < L-1]
        k_node<<<nodeBlocks, 256, 0, stream>>>(h, nmsg, WoPl,
                                               bo + (size_t)l * HH,
                                               bn_g + (size_t)l * HH,
                                               bn_b + (size_t)l * HH,
                                               (l < LL - 1) ? 1 : 0, h);
    }

    // mean pool over graphs
    k_clear<<<(GG * 128 + 255) / 256, 256, 0, stream>>>(sums, GG * 128);
    k_clear<<<(GG + 255) / 256, 256, 0, stream>>>(cnts, GG);
    k_pool<<<(poolN + 255) / 256, 256, 0, stream>>>(h, batch, sums, cnts, poolN);
    k_div<<<(GG * 128 + 255) / 256, 256, 0, stream>>>(sums, cnts, gemb);

    // classifier
    k_cls1<<<GG / 16, 128, 0, stream>>>(gemb, W1P, cls_b1, hid);
    k_cls2<<<(GG * 2 + 255) / 256, 256, 0, stream>>>(hid, cls_W2, cls_b2, out);
}
